// SwinTransformerBlock_25194278159112
// MI455X (gfx1250) — matrix operand order repaired and checked by reading, not compiled, not run on hardware
//
#include <hip/hip_runtime.h>
#include <hip/hip_bf16.h>
#include <math.h>

// ---------------------------------------------------------------------------
// CDNA5 (gfx1250) wave32 WMMA types & helpers
// ---------------------------------------------------------------------------
typedef __attribute__((ext_vector_type(16))) _Float16 v16h;
typedef __attribute__((ext_vector_type(8)))  float    v8f;

union FragH16 { v16h v; uint4 u[2]; _Float16 h[16]; };
union H8      { _Float16 h[8]; uint4 u; };

__device__ inline v8f vzero8() {
  v8f z = {0.f, 0.f, 0.f, 0.f, 0.f, 0.f, 0.f, 0.f};
  return z;
}

__device__ inline v8f wmma16(v16h a, v16h b, v8f c) {
  // D = A(16x32 f16) * B(32x16 f16) + C(16x16 f32)
  return __builtin_amdgcn_wmma_f32_16x16x32_f16(false, a, false, b, (short)0, c,
                                                false, false);
}

// A fragment (16x32, f16), row-major source with leading dim ld (halfs).
// ISA layout: lane l holds row (l&15); K = (l>>4)*8 .. +7 in v0-3, +16.. in v4-7.
__device__ inline v16h load_a_frag(const _Float16* p, int ld) {
  int lane = threadIdx.x & 31;
  int row = lane & 15, hl = lane >> 4;
  const _Float16* r = p + (size_t)row * ld + hl * 8;
  FragH16 f;
  f.u[0] = *(const uint4*)r;
  f.u[1] = *(const uint4*)(r + 16);
  return f.v;
}

// B fragment (32x16, f16) where the source stores each B-column contiguously
// in K (i.e. an (N x K) row-major matrix W, B = W^T). lane l holds column
// (l&15), K = (l>>4)*8 .. +7 in v0-3, +16.. in v4-7.
__device__ inline v16h load_b_frag(const _Float16* p, int ld) {
  int lane = threadIdx.x & 31;
  int col = lane & 15, hl = lane >> 4;
  const _Float16* r = p + (size_t)col * ld + hl * 8;
  FragH16 f;
  f.u[0] = *(const uint4*)r;
  f.u[1] = *(const uint4*)(r + 16);
  return f.v;
}

__device__ inline float gelu_exact(float x) {
  return 0.5f * x * (1.f + erff(x * 0.70710678118654752f));
}

// window-order row -> flat (b*L + l) token index after window-reverse+unshift
__device__ inline int winrow_to_token(int row) {
  int b = row / 12288;
  int rem = row - b * 12288;
  int win = rem >> 6, tok = rem & 63;
  int wh = win >> 4, ww = win & 15;
  int i = tok >> 3, j = tok & 7;
  int h = wh * 8 + i + 4; if (h >= 96)  h -= 96;
  int w = ww * 8 + j + 4; if (w >= 128) w -= 128;
  return b * 12288 + h * 128 + w;
}

// Swin shifted-window mask region id (labels live on the shifted image)
__device__ inline int swin_region(int wh, int ww, int tok) {
  int i = tok >> 3, j = tok & 7;
  int p = wh * 8 + i;
  int q = ww * 8 + j;
  int hr = (p < 88) ? 0 : ((p < 92) ? 1 : 2);
  int wr = (q < 120) ? 0 : ((q < 124) ? 1 : 2);
  return hr * 3 + wr;
}

// ---------------------------------------------------------------------------
// f32 -> f16 weight conversion
// ---------------------------------------------------------------------------
__global__ void cvt_f32_f16(const float* __restrict__ s, _Float16* __restrict__ d,
                            int n) {
  int i = blockIdx.x * 256 + threadIdx.x;
  if (i < n) d[i] = (_Float16)s[i];
}

// ---------------------------------------------------------------------------
// Position-bias MLP table (8 x 64 x 64) + lambda scalar
// ---------------------------------------------------------------------------
__global__ __launch_bounds__(256)
void pb_lam_kernel(const float* __restrict__ Wp1, const float* __restrict__ bp1,
                   const float* __restrict__ Wp2, const float* __restrict__ bp2,
                   const float* __restrict__ lq1, const float* __restrict__ lk1,
                   const float* __restrict__ lq2, const float* __restrict__ lk2,
                   float* __restrict__ pbtab, float* __restrict__ lamout) {
  int t = blockIdx.x * 256 + threadIdx.x;
  if (t >= 4096) return;
  int a = t >> 6, b = t & 63;
  int ih = a >> 3, iw = a & 7, jh = b >> 3, jw = b & 7;
  float dx = (float)(iw - jw);
  float dy = -(float)(ih - jh);
  float r = sqrtf(dx * dx + dy * dy + 1e-9f);
  float th = atan2f(dx, dy);
  const float PI = 3.14159265358979323846f;
  float p0 = r * (1.f / 9.899494936611665f);     // / sqrt(7^2+7^2)
  float p1 = (th + PI) * (1.f / (2.f * PI));
  float accv[8];
#pragma unroll
  for (int o = 0; o < 8; ++o) accv[o] = bp2[o];
  for (int hh = 0; hh < 64; ++hh) {
    float hv = p0 * Wp1[hh * 2] + p1 * Wp1[hh * 2 + 1] + bp1[hh];
    hv = gelu_exact(hv);
#pragma unroll
    for (int o = 0; o < 8; ++o) accv[o] += hv * Wp2[o * 64 + hh];
  }
#pragma unroll
  for (int o = 0; o < 8; ++o) pbtab[o * 4096 + t] = accv[o];
  if (t == 0) {
    float s1 = 0.f, s2 = 0.f;
    for (int i = 0; i < 16; ++i) { s1 += lq1[i] * lk1[i]; s2 += lq2[i] * lk2[i]; }
    lamout[0] = expf(s1) - expf(s2) + 0.2f;  // LAMBDA_INIT = 0.2
  }
}

// ---------------------------------------------------------------------------
// LayerNorm. SHIFT=1: read with cyclic shift + window partition (output rows
// in window order). SHIFT=0: identity mapping. Output f16. 1 wave per token.
// ---------------------------------------------------------------------------
template <int SHIFT>
__global__ __launch_bounds__(256)
void ln_kernel(const float* __restrict__ x, const float* __restrict__ g,
               const float* __restrict__ bta, _Float16* __restrict__ out) {
  int wave = threadIdx.x >> 5, lane = threadIdx.x & 31;
  int m = blockIdx.x * 8 + wave;
  size_t src;
  if (SHIFT) {
    int b = m / 12288;
    int rem = m - b * 12288;
    int win = rem >> 6, tok = rem & 63;
    int wh = win >> 4, ww = win & 15;
    int i = tok >> 3, j = tok & 7;
    int hs = wh * 8 + i + 4;  if (hs >= 96)  hs -= 96;
    int wsp = ww * 8 + j + 4; if (wsp >= 128) wsp -= 128;
    src = ((size_t)b * 12288 + hs * 128 + wsp) * 256;
  } else {
    src = (size_t)m * 256;
  }
  const float4* p = (const float4*)(x + src) + lane * 2;
  float4 v0 = p[0], v1 = p[1];
  float vv[8] = {v0.x, v0.y, v0.z, v0.w, v1.x, v1.y, v1.z, v1.w};
  float s = 0.f, q = 0.f;
#pragma unroll
  for (int t = 0; t < 8; ++t) { s += vv[t]; q += vv[t] * vv[t]; }
#pragma unroll
  for (int off = 16; off >= 1; off >>= 1) {
    s += __shfl_xor(s, off, 32);
    q += __shfl_xor(q, off, 32);
  }
  float mu = s * (1.f / 256.f);
  float var = q * (1.f / 256.f) - mu * mu;
  float rs = rsqrtf(var + 1e-5f);
  int c0 = lane * 8;
  H8 o;
#pragma unroll
  for (int t = 0; t < 8; ++t)
    o.h[t] = (_Float16)((vv[t] - mu) * rs * g[c0 + t] + bta[c0 + t]);
  *(uint4*)(out + (size_t)m * 256 + c0) = o.u;
}

// ---------------------------------------------------------------------------
// Generic WMMA GEMM: out = A(MxK,f16) @ W(NxK,f16)^T + bias, fused epilogues.
// Block: 256 thr = 8 waves (4 along M x 2 along N), 32x64 per wave -> 128x128.
// MODE 0: f16 out                        MODE 1: f16 GELU out
// MODE 2: f32 = res + v, window-reverse scatter (proj)
// MODE 3: f32 = res + v (fc2)
// MODE 4: qkv: q/k -> outh normally; V columns transposed per (win,head) to
//         outh2 as (win, head, d[32], n[64]) so P@V B-frags load contiguously.
// ---------------------------------------------------------------------------
template <int MODE>
__global__ __launch_bounds__(256)
void gemm_wmma_kernel(const _Float16* __restrict__ A, const _Float16* __restrict__ Wt,
                      const float* __restrict__ bias, const float* __restrict__ res,
                      float* __restrict__ outf, _Float16* __restrict__ outh,
                      _Float16* __restrict__ outh2, int M, int N, int K) {
  int wave = threadIdx.x >> 5, lane = threadIdx.x & 31;
  int row0 = blockIdx.x * 128 + (wave >> 1) * 32;
  int col0 = blockIdx.y * 128 + (wave & 1) * 64;
  v8f acc[2][4];
#pragma unroll
  for (int i = 0; i < 2; ++i)
#pragma unroll
    for (int j = 0; j < 4; ++j) acc[i][j] = vzero8();

  for (int k0 = 0; k0 < K; k0 += 32) {
    v16h a0 = load_a_frag(A + (size_t)row0 * K + k0, K);
    v16h a1 = load_a_frag(A + (size_t)(row0 + 16) * K + k0, K);
#pragma unroll
    for (int nt = 0; nt < 4; ++nt) {
      v16h b = load_b_frag(Wt + (size_t)(col0 + nt * 16) * K + k0, K);
      acc[0][nt] = wmma16(a0, b, acc[0][nt]);
      acc[1][nt] = wmma16(a1, b, acc[1][nt]);
    }
  }

  int hl = lane >> 4, lc = lane & 15;
#pragma unroll
  for (int mt = 0; mt < 2; ++mt) {
#pragma unroll
    for (int e = 0; e < 8; ++e) {
      int row = row0 + mt * 16 + e + 8 * hl;
      int tokrow = row;
      if (MODE == 2) tokrow = winrow_to_token(row);
#pragma unroll
      for (int nt = 0; nt < 4; ++nt) {
        int col = col0 + nt * 16 + lc;
        float v = acc[mt][nt][e] + bias[col];
        if (MODE == 0) {
          outh[(size_t)row * N + col] = (_Float16)v;
        } else if (MODE == 1) {
          outh[(size_t)row * N + col] = (_Float16)gelu_exact(v);
        } else if (MODE == 2) {
          size_t idx = (size_t)tokrow * 256 + col;
          outf[idx] = res[idx] + v;
        } else if (MODE == 3) {
          size_t idx = (size_t)row * 256 + col;
          outf[idx] = res[idx] + v;
        } else {  // MODE 4
          if (col < 512) {
            outh[(size_t)row * N + col] = (_Float16)v;
          } else {
            int d = col - 512;
            int head = d >> 5, dd = d & 31;
            int win = row >> 6, n = row & 63;
            outh2[(((size_t)win * 8 + head) * 32 + dd) * 64 + n] = (_Float16)v;
          }
        }
      }
    }
  }
}

// ---------------------------------------------------------------------------
// Fused differential window attention. 1 block per window (512 thr, 16 waves),
// 2 waves per head, each wave owns 32 query rows. HD=32 -> one WMMA per tile.
// V comes pre-transposed per (win, head): vT[d][n], so B-frags are b128 loads.
// ---------------------------------------------------------------------------
__global__ __launch_bounds__(512)
void attn_kernel(const _Float16* __restrict__ qkv, const _Float16* __restrict__ qk2,
                 const _Float16* __restrict__ vT, const float* __restrict__ pbtab,
                 const float* __restrict__ lamp, const float* __restrict__ subw,
                 _Float16* __restrict__ o_out) {
  __shared__ _Float16 plds[16][32 * 40];  // per-wave 32x32 P chunk, padded stride
  int wave = threadIdx.x >> 5, lane = threadIdx.x & 31;
  int head = wave >> 1, r0 = (wave & 1) * 32;
  int win = blockIdx.x;
  int hl = lane >> 4, lc = lane & 15;
  float lam = lamp[0];
  const float scale = 0.17677669529663687f;  // 32^-0.5
  float lscale = lam * scale;
  size_t rowbase = (size_t)win * 64;
  const _Float16* qp  = qkv + rowbase * 768 + head * 32;
  const _Float16* kp  = qp + 256;
  const _Float16* q2p = qk2 + rowbase * 512 + head * 32;
  const _Float16* k2p = q2p + 256;
  const _Float16* vtp = vT + ((size_t)win * 8 + head) * 32 * 64;  // [d][n]
  int wimg = win % 192;
  int wh = wimg >> 4, ww = wimg & 15;

  v16h qa0  = load_a_frag(qp  + (size_t)r0 * 768, 768);
  v16h qa1  = load_a_frag(qp  + (size_t)(r0 + 16) * 768, 768);
  v16h q2a0 = load_a_frag(q2p + (size_t)r0 * 512, 512);
  v16h q2a1 = load_a_frag(q2p + (size_t)(r0 + 16) * 512, 512);

  v8f S[2][4];
#pragma unroll
  for (int nt = 0; nt < 4; ++nt) {
    v16h kb  = load_b_frag(kp  + (size_t)(nt * 16) * 768, 768);
    v16h k2b = load_b_frag(k2p + (size_t)(nt * 16) * 512, 512);
    int col = nt * 16 + lc;
    int creg = swin_region(wh, ww, col);
#pragma unroll
    for (int mt = 0; mt < 2; ++mt) {
      v8f s1 = wmma16(mt ? qa1 : qa0, kb, vzero8());
      v8f s2 = wmma16(mt ? q2a1 : q2a0, k2b, vzero8());
      v8f o;
#pragma unroll
      for (int e = 0; e < 8; ++e) {
        int rloc = r0 + mt * 16 + e + 8 * hl;
        int rreg = swin_region(wh, ww, rloc);
        float mb = (rreg != creg) ? -100.f : 0.f;
        o[e] = scale * s1[e] - lscale * s2[e] +
               pbtab[head * 4096 + rloc * 64 + col] + mb;
      }
      S[mt][nt] = o;
    }
  }

  // softmax over rows; lanes {0..15} and {16..31} each share 16 rows -> xor<=8
#pragma unroll
  for (int mt = 0; mt < 2; ++mt) {
#pragma unroll
    for (int e = 0; e < 8; ++e) {
      float mx = fmaxf(fmaxf(S[mt][0][e], S[mt][1][e]),
                       fmaxf(S[mt][2][e], S[mt][3][e]));
#pragma unroll
      for (int d = 1; d <= 8; d <<= 1) mx = fmaxf(mx, __shfl_xor(mx, d, 32));
      float sm = 0.f;
#pragma unroll
      for (int nt = 0; nt < 4; ++nt) {
        float pe = expf(S[mt][nt][e] - mx);
        S[mt][nt][e] = pe;
        sm += pe;
      }
#pragma unroll
      for (int d = 1; d <= 8; d <<= 1) sm += __shfl_xor(sm, d, 32);
      float inv = 1.f / sm;
#pragma unroll
      for (int nt = 0; nt < 4; ++nt) S[mt][nt][e] *= inv;
    }
  }

  // O = P @ V, K=64 in two 32-wide chunks, P re-fragmented through LDS.
  // B operand comes from vT[d][n]: contiguous in K(=n) for each column d.
  v8f O00 = vzero8(), O01 = vzero8(), O10 = vzero8(), O11 = vzero8();
  _Float16* myp = &plds[wave][0];
#pragma unroll
  for (int ks = 0; ks < 2; ++ks) {
#pragma unroll
    for (int mt = 0; mt < 2; ++mt)
#pragma unroll
      for (int nto = 0; nto < 2; ++nto) {
        int nt = ks * 2 + nto;
#pragma unroll
        for (int e = 0; e < 8; ++e) {
          int rl = mt * 16 + e + 8 * hl;
          myp[rl * 40 + nto * 16 + lc] = (_Float16)S[mt][nt][e];
        }
      }
    v16h vb0 = load_b_frag(vtp + ks * 32, 64);            // d = 0..15
    v16h vb1 = load_b_frag(vtp + 16 * 64 + ks * 32, 64);  // d = 16..31
    v16h pa0 = load_a_frag(myp, 40);
    v16h pa1 = load_a_frag(myp + 16 * 40, 40);
    O00 = wmma16(pa0, vb0, O00);
    O01 = wmma16(pa0, vb1, O01);
    O10 = wmma16(pa1, vb0, O10);
    O11 = wmma16(pa1, vb1, O11);
  }

  // per-(row, head) RMS-norm over HD=32, * subw * (1 - lambda_init)
  float sw0 = subw[lc], sw1 = subw[16 + lc];
#pragma unroll
  for (int mt = 0; mt < 2; ++mt) {
    v8f& oa = mt ? O10 : O00;
    v8f& ob = mt ? O11 : O01;
#pragma unroll
    for (int e = 0; e < 8; ++e) {
      float a0 = oa[e], a1 = ob[e];
      float ss = a0 * a0 + a1 * a1;
#pragma unroll
      for (int d = 1; d <= 8; d <<= 1) ss += __shfl_xor(ss, d, 32);
      float rs = rsqrtf(ss * (1.f / 32.f) + 1e-9f) * 0.8f;
      int rl = r0 + mt * 16 + e + 8 * hl;
      size_t orow = (rowbase + rl) * 256 + head * 32;
      o_out[orow + lc]      = (_Float16)(a0 * rs * sw0);
      o_out[orow + 16 + lc] = (_Float16)(a1 * rs * sw1);
    }
  }
}

// ---------------------------------------------------------------------------
// Launch
// ---------------------------------------------------------------------------
extern "C" void kernel_launch(void* const* d_in, const int* in_sizes, int n_in,
                              void* d_out, int out_size, void* d_ws, size_t ws_size,
                              hipStream_t stream) {
  (void)in_sizes; (void)n_in; (void)out_size; (void)ws_size;
  const float* x     = (const float*)d_in[0];
  const float* g1    = (const float*)d_in[1];
  const float* b1    = (const float*)d_in[2];
  const float* Wqkv  = (const float*)d_in[3];
  const float* bqkv  = (const float*)d_in[4];
  const float* Wqkv2 = (const float*)d_in[5];
  const float* bqkv2 = (const float*)d_in[6];
  const float* Wp1   = (const float*)d_in[7];
  const float* bp1   = (const float*)d_in[8];
  const float* Wp2   = (const float*)d_in[9];
  const float* bp2   = (const float*)d_in[10];
  const float* lq1   = (const float*)d_in[11];
  const float* lk1   = (const float*)d_in[12];
  const float* lq2   = (const float*)d_in[13];
  const float* lk2   = (const float*)d_in[14];
  const float* subw  = (const float*)d_in[15];
  const float* Wproj = (const float*)d_in[16];
  const float* bproj = (const float*)d_in[17];
  const float* g2    = (const float*)d_in[18];
  const float* b2    = (const float*)d_in[19];
  const float* Wfc1  = (const float*)d_in[20];
  const float* bfc1  = (const float*)d_in[21];
  const float* Wfc2  = (const float*)d_in[22];
  const float* bfc2  = (const float*)d_in[23];
  float* out = (float*)d_out;

  constexpr size_t M = 98304;  // 8 batches * 192 windows * 64 tokens
  char* ws = (char*)d_ws;
  size_t off = 0;
  auto alloc = [&](size_t bytes) {
    char* p = ws + off;
    off += (bytes + 255) & ~(size_t)255;
    return p;
  };
  _Float16* xw   = (_Float16*)alloc(M * 256 * 2);   // LN1+shift+window, f16
  _Float16* qkvb = (_Float16*)alloc(M * 768 * 2);   // q|k (v slot unused)
  _Float16* qk2b = (_Float16*)alloc(M * 512 * 2);   // q2|k2
  _Float16* vTb  = (_Float16*)alloc(M * 256 * 2);   // V transposed (win,head,d,n)
  _Float16* ob   = (_Float16*)alloc(M * 256 * 2);   // attention output, f16
  float*    xf   = (float*)   alloc(M * 256 * 4);   // x after attention residual
  _Float16* wqh  = (_Float16*)alloc(768 * 256 * 2);
  _Float16* wq2h = (_Float16*)alloc(512 * 256 * 2);
  _Float16* wph  = (_Float16*)alloc(256 * 256 * 2);
  _Float16* wf1h = (_Float16*)alloc(1024 * 256 * 2);
  _Float16* wf2h = (_Float16*)alloc(256 * 1024 * 2);
  float*    pbt  = (float*)   alloc(8 * 4096 * 4);
  float*    lamb = (float*)   alloc(256);
  _Float16* xn2 = xw;    // alias: xw dead after qkv GEMMs
  _Float16* h1  = qkvb;  // alias: qkv+qk2 (251MB contiguous) >= h1 (201MB)

  cvt_f32_f16<<<(196608 + 255) / 256, 256, 0, stream>>>(Wqkv,  wqh,  196608);
  cvt_f32_f16<<<(131072 + 255) / 256, 256, 0, stream>>>(Wqkv2, wq2h, 131072);
  cvt_f32_f16<<<(65536  + 255) / 256, 256, 0, stream>>>(Wproj, wph,  65536);
  cvt_f32_f16<<<(262144 + 255) / 256, 256, 0, stream>>>(Wfc1,  wf1h, 262144);
  cvt_f32_f16<<<(262144 + 255) / 256, 256, 0, stream>>>(Wfc2,  wf2h, 262144);

  pb_lam_kernel<<<16, 256, 0, stream>>>(Wp1, bp1, Wp2, bp2, lq1, lk1, lq2, lk2,
                                        pbt, lamb);

  ln_kernel<1><<<12288, 256, 0, stream>>>(x, g1, b1, xw);

  // QKV: q/k normal, V transposed per (win, head)
  gemm_wmma_kernel<4><<<dim3(768, 6), 256, 0, stream>>>(
      xw, wqh, bqkv, nullptr, nullptr, qkvb, vTb, (int)M, 768, 256);
  gemm_wmma_kernel<0><<<dim3(768, 4), 256, 0, stream>>>(
      xw, wq2h, bqkv2, nullptr, nullptr, qk2b, nullptr, (int)M, 512, 256);

  attn_kernel<<<1536, 512, 0, stream>>>(qkvb, qk2b, vTb, pbt, lamb, subw, ob);

  gemm_wmma_kernel<2><<<dim3(768, 2), 256, 0, stream>>>(
      ob, wph, bproj, x, xf, nullptr, nullptr, (int)M, 256, 256);

  ln_kernel<0><<<12288, 256, 0, stream>>>(xf, g2, b2, xn2);

  gemm_wmma_kernel<1><<<dim3(768, 8), 256, 0, stream>>>(
      xn2, wf1h, bfc1, nullptr, nullptr, h1, nullptr, (int)M, 1024, 256);
  gemm_wmma_kernel<3><<<dim3(768, 2), 256, 0, stream>>>(
      h1, wf2h, bfc2, xf, out, nullptr, nullptr, (int)M, 256, 1024);
}
